// SelfAttention_55972013802083
// MI455X (gfx1250) — matrix operand order repaired and checked by reading, not compiled, not run on hardware
//
#include <hip/hip_runtime.h>

// ---------------------------------------------------------------------------
// CDNA5 (gfx1250, wave32) fused causal self-attention:
//   qkv = x @ w_qkv ; flash-attention per head ; out = attn @ w_proj
// Matrix math via v_wmma_f32_16x16x32_bf16 (fp32 accumulate);
// A/Q/K tiles staged with global_load_async_to_lds_b128 (ASYNCcnt DMA).
// ---------------------------------------------------------------------------

typedef __attribute__((ext_vector_type(16))) __bf16 v16bf;
typedef __attribute__((ext_vector_type(8)))  __bf16 v8bf;
typedef __attribute__((ext_vector_type(4)))  __bf16 v4bf;
typedef __attribute__((ext_vector_type(8)))  float  v8f;

#define BSZ   4
#define TSEQ  2048
#define NEMBD 1024
#define NHEAD 16
#define HDIM  64

__device__ __forceinline__ v8f wmma_bf16(v16bf a, v16bf b, v8f c) {
  return __builtin_amdgcn_wmma_f32_16x16x32_bf16(false, a, false, b,
                                                 (short)0, c, false, false);
}

// Async DMA: 16B global -> LDS, tracked by ASYNCcnt.
// LDS byte offset = low 32 bits of the generic pointer (LDS aperture encoding).
__device__ __forceinline__ void async_copy_b128(__bf16* lds_dst,
                                                const __bf16* gsrc) {
  unsigned loff = (unsigned)(size_t)lds_dst;
  asm volatile("global_load_async_to_lds_b128 %0, %1, off"
               :: "v"(loff), "v"((unsigned long long)(size_t)gsrc)
               : "memory");
}
__device__ __forceinline__ void wait_async() {
  asm volatile("s_wait_asynccnt 0x0" ::: "memory");
}

// A-fragment (16x32 bf16, row-major source, `stride` elements per row).
// ISA layout: lane m=lane&15, half=lane>>4; elem i -> k=(i/8)*16 + half*8 + i%8
__device__ __forceinline__ v16bf load_frag_a(const __bf16* base, int stride) {
  const int lane = threadIdx.x & 31;
  const int m = lane & 15, hf = lane >> 4;
  const v8bf lo = *(const v8bf*)(base + m * stride + hf * 8);
  const v8bf hi = *(const v8bf*)(base + m * stride + 16 + hf * 8);
  v16bf r;
#pragma unroll
  for (int i = 0; i < 8; ++i) { r[i] = lo[i]; r[i + 8] = hi[i]; }
  return r;
}

// B-fragment (32x16 bf16). Source stored K-contiguous per column:
// base[n*stride + k], n=lane&15, elem i -> k=(i/8)*16 + half*8 + i%8.
__device__ __forceinline__ v16bf load_frag_b(const __bf16* base, int stride) {
  const int lane = threadIdx.x & 31;
  const int n = lane & 15, hf = lane >> 4;
  const v8bf lo = *(const v8bf*)(base + n * stride + hf * 8);
  const v8bf hi = *(const v8bf*)(base + n * stride + 16 + hf * 8);
  v16bf r;
#pragma unroll
  for (int i = 0; i < 8; ++i) { r[i] = lo[i]; r[i + 8] = hi[i]; }
  return r;
}

// ---------------------------------------------------------------------------
// fp32 -> bf16 downconvert (n multiple of 4)
// ---------------------------------------------------------------------------
__global__ void cvt_f32_bf16(const float* __restrict__ in,
                             __bf16* __restrict__ out, int n) {
  int i = (blockIdx.x * blockDim.x + threadIdx.x) * 4;
  if (i >= n) return;
  const float4 f = *(const float4*)(in + i);
  v4bf o;
  o[0] = (__bf16)f.x; o[1] = (__bf16)f.y; o[2] = (__bf16)f.z; o[3] = (__bf16)f.w;
  *(v4bf*)(out + i) = o;
}

// ---------------------------------------------------------------------------
// Shared GEMM mainloop: C(128x128) += A(128xK) * B(KxN) tile at (m0,n0).
// 256 threads = 8 waves; wave (wm, wn) owns a 32x64 sub-tile (2x4 WMMA
// accumulators, 8 WMMA per K-step -> ~1.5 LDS b128 loads per WMMA).
// sA: [128][32] bf16 row-major (async-DMA staged);
// sBt: [128][32] bf16 (B staged transposed -> contiguous B-fragment reads).
// ---------------------------------------------------------------------------
__device__ __forceinline__ void gemm_mainloop(
    const __bf16* __restrict__ A, const __bf16* __restrict__ B,
    int N, int K, int m0, int n0,
    __bf16* sA, __bf16* sBt, v8f acc[2][4]) {
  const int tid = threadIdx.x;
  const int wid = tid >> 5;
  const int wm = wid & 3, wn = wid >> 2;

  for (int kb = 0; kb < K; kb += 32) {
    // stage A tile 128x32 via async DMA (512 16B chunks / 256 threads)
#pragma unroll
    for (int j = 0; j < 2; ++j) {
      int c = j * 256 + tid;
      int row = c >> 2, off = (c & 3) * 8;
      async_copy_b128(sA + row * 32 + off,
                      A + (size_t)(m0 + row) * K + kb + off);
    }
    // stage B tile 32x128 transposed into sBt[128][32]
#pragma unroll
    for (int j = 0; j < 2; ++j) {
      int c = j * 256 + tid;
      int k = c >> 4, off = (c & 15) * 8;
      v8bf bv = *(const v8bf*)(B + (size_t)(kb + k) * N + n0 + off);
#pragma unroll
      for (int e = 0; e < 8; ++e) sBt[(off + e) * 32 + k] = bv[e];
    }
    if (kb + 32 < K)
      __builtin_prefetch(
          B + (size_t)(kb + 32 + (tid >> 4)) * N + n0 + (tid & 15) * 8, 0, 3);
    wait_async();
    __syncthreads();

    v16bf a0 = load_frag_a(sA + (wm * 32 + 0) * 32, 32);
    v16bf a1 = load_frag_a(sA + (wm * 32 + 16) * 32, 32);
#pragma unroll
    for (int nt = 0; nt < 4; ++nt) {
      v16bf bfr = load_frag_b(sBt + (wn * 64 + nt * 16) * 32, 32);
      acc[0][nt] = wmma_bf16(a0, bfr, acc[0][nt]);
      acc[1][nt] = wmma_bf16(a1, bfr, acc[1][nt]);
    }
    __syncthreads();
  }
}

// ---------------------------------------------------------------------------
// QKV GEMM: X[8192,1024] * Wqkv[1024,3072]; scatter bf16 into head-major
// Q/K/V [B,H,T,D].  grid (64, 24), 256 threads.
// ---------------------------------------------------------------------------
__global__ void qkv_gemm(const __bf16* __restrict__ Xb,
                         const __bf16* __restrict__ Wb,
                         __bf16* __restrict__ Qw, __bf16* __restrict__ Kw,
                         __bf16* __restrict__ Vw) {
  __shared__ __bf16 sA[128 * 32];
  __shared__ __bf16 sBt[128 * 32];
  v8f z = {0.f, 0.f, 0.f, 0.f, 0.f, 0.f, 0.f, 0.f};
  v8f acc[2][4] = {{z, z, z, z}, {z, z, z, z}};

  const int m0 = blockIdx.x * 128, n0 = blockIdx.y * 128;
  gemm_mainloop(Xb, Wb, 3 * NEMBD, NEMBD, m0, n0, sA, sBt, acc);

  const int lane = threadIdx.x & 31, ln = lane & 15, hf = lane >> 4;
  const int wid = threadIdx.x >> 5, wm = wid & 3, wn = wid >> 2;
#pragma unroll
  for (int mt = 0; mt < 2; ++mt)
#pragma unroll
    for (int nt = 0; nt < 4; ++nt)
#pragma unroll
      for (int v = 0; v < 8; ++v) {
        int R = m0 + wm * 32 + mt * 16 + v + hf * 8;
        int Cc = n0 + wn * 64 + nt * 16 + ln;
        int mat = Cc >> 10, wi = Cc & 1023;
        int h = wi >> 6, d = wi & 63;
        int b = R >> 11, t = R & 2047;
        size_t idx = (((size_t)b * NHEAD + h) * TSEQ + t) * HDIM + d;
        __bf16 val = (__bf16)acc[mt][nt][v];
        (mat == 0 ? Qw : (mat == 1 ? Kw : Vw))[idx] = val;
      }
}

// ---------------------------------------------------------------------------
// Flash attention: grid (T/64, B*H), 128 threads = 4 waves.
// Wave w owns 16 query rows; 32-key blocks with online softmax; logits never
// leave the WGP.  Q/K tiles async-DMA staged; V staged transposed.
// Writes attn-out bf16 in [B,T,H,D] (== [B,T,C]).
// ---------------------------------------------------------------------------
__global__ void attn_kernel(const __bf16* __restrict__ Q,
                            const __bf16* __restrict__ K,
                            const __bf16* __restrict__ V,
                            __bf16* __restrict__ AO) {
  __shared__ __bf16 Qs[64 * 64];   // 64 queries x 64 dims
  __shared__ __bf16 Ks[32 * 64];   // 32 keys x 64 dims (row-major)
  __shared__ __bf16 Vts[64 * 32];  // V transposed: [dim][key]
  __shared__ __bf16 Ps[4 * 16 * 32];  // per-wave P staging (C-layout -> A-layout)

  const int tid = threadIdx.x;
  const int w = tid >> 5;
  const int lane = tid & 31, ln = lane & 15, hf = lane >> 4;
  const int q0 = blockIdx.x * 64;
  const int bh = blockIdx.y;
  const size_t base = (size_t)bh * TSEQ * HDIM;

  // stage Q tile via async DMA (512 16B chunks / 128 threads)
#pragma unroll
  for (int j = 0; j < 4; ++j) {
    int c = j * 128 + tid;
    int row = c >> 3, off = (c & 7) * 8;
    async_copy_b128(Qs + row * 64 + off,
                    Q + base + (size_t)(q0 + row) * HDIM + off);
  }
  wait_async();
  __syncthreads();

  const v16bf aq0 = load_frag_a(Qs + (w * 16) * 64 + 0, 64);
  const v16bf aq1 = load_frag_a(Qs + (w * 16) * 64 + 32, 64);

  v8f z = {0.f, 0.f, 0.f, 0.f, 0.f, 0.f, 0.f, 0.f};
  v8f o[4] = {z, z, z, z};
  float mrow[8], lrow[8];
#pragma unroll
  for (int v = 0; v < 8; ++v) { mrow[v] = -1e30f; lrow[v] = 0.f; }
  const float scale = 0.125f;  // 1/sqrt(64)

  const int nkb = (q0 >> 5) + 2;  // causal: key blocks up to q0+63
  for (int ib = 0; ib < nkb; ++ib) {
    const int kb = ib * 32;
    // stage K block (async, row-major) and V block (transposed via VGPRs)
#pragma unroll
    for (int j = 0; j < 2; ++j) {
      int c = j * 128 + tid;
      int row = c >> 3, off = (c & 7) * 8;
      async_copy_b128(Ks + row * 64 + off,
                      K + base + (size_t)(kb + row) * HDIM + off);
      v8bf vv = *(const v8bf*)(V + base + (size_t)(kb + row) * HDIM + off);
#pragma unroll
      for (int e = 0; e < 8; ++e) Vts[(off + e) * 32 + row] = vv[e];
    }
    wait_async();
    __syncthreads();

    // S = Q(16x64) . K^T(64x32): 2 N-tiles x 2 K-slices
    v8f s0 = z, s1 = z;
    s0 = wmma_bf16(aq0, load_frag_b(Ks + 0 * 64 + 0, 64), s0);
    s0 = wmma_bf16(aq1, load_frag_b(Ks + 0 * 64 + 32, 64), s0);
    s1 = wmma_bf16(aq0, load_frag_b(Ks + 16 * 64 + 0, 64), s1);
    s1 = wmma_bf16(aq1, load_frag_b(Ks + 16 * 64 + 32, 64), s1);

    // causal mask + scale + online softmax (rows replicated across 16-lane half)
#pragma unroll
    for (int v = 0; v < 8; ++v) {
      const int qrow = q0 + w * 16 + v + hf * 8;
      const int k0 = kb + ln, k1 = kb + 16 + ln;
      float x0 = (k0 <= qrow) ? s0[v] * scale : -1e30f;
      float x1 = (k1 <= qrow) ? s1[v] * scale : -1e30f;
      float mv = fmaxf(x0, x1);
#pragma unroll
      for (int off = 1; off < 16; off <<= 1)
        mv = fmaxf(mv, __shfl_xor(mv, off, 32));
      const float mnew = fmaxf(mrow[v], mv);
      const float alpha = __expf(mrow[v] - mnew);
      const float p0 = __expf(x0 - mnew);
      const float p1 = __expf(x1 - mnew);
      float rs = p0 + p1;
#pragma unroll
      for (int off = 1; off < 16; off <<= 1) rs += __shfl_xor(rs, off, 32);
      lrow[v] = lrow[v] * alpha + rs;
      mrow[v] = mnew;
      o[0][v] *= alpha; o[1][v] *= alpha; o[2][v] *= alpha; o[3][v] *= alpha;
      // stage P (C-layout) into per-wave LDS for A-fragment reload
      __bf16* st = Ps + w * 512 + (v + hf * 8) * 32;
      st[ln] = (__bf16)p0;
      st[16 + ln] = (__bf16)p1;
    }

    // O += P(16x32) . V(32x64): one A-fragment, 4 N-tiles
    const v16bf aP = load_frag_a(Ps + w * 512, 32);
    o[0] = wmma_bf16(aP, load_frag_b(Vts + 0 * 32, 32), o[0]);
    o[1] = wmma_bf16(aP, load_frag_b(Vts + 16 * 32, 32), o[1]);
    o[2] = wmma_bf16(aP, load_frag_b(Vts + 32 * 32, 32), o[2]);
    o[3] = wmma_bf16(aP, load_frag_b(Vts + 48 * 32, 32), o[3]);
    __syncthreads();
  }

  // normalize + store to [B,T,H,D]
  const int b = bh >> 4, h = bh & 15;
#pragma unroll
  for (int dt = 0; dt < 4; ++dt)
#pragma unroll
    for (int v = 0; v < 8; ++v) {
      const int qrow = q0 + w * 16 + v + hf * 8;
      const float val = o[dt][v] / lrow[v];
      size_t idx = (((size_t)b * TSEQ + qrow) * NHEAD + h) * HDIM + dt * 16 + ln;
      AO[idx] = (__bf16)val;
    }
}

// ---------------------------------------------------------------------------
// Output projection: AO[8192,1024] * Wproj[1024,1024] -> fp32 d_out
// grid (64, 8), 256 threads.
// ---------------------------------------------------------------------------
__global__ void proj_gemm(const __bf16* __restrict__ Ab,
                          const __bf16* __restrict__ Wb,
                          float* __restrict__ Out) {
  __shared__ __bf16 sA[128 * 32];
  __shared__ __bf16 sBt[128 * 32];
  v8f z = {0.f, 0.f, 0.f, 0.f, 0.f, 0.f, 0.f, 0.f};
  v8f acc[2][4] = {{z, z, z, z}, {z, z, z, z}};

  const int m0 = blockIdx.x * 128, n0 = blockIdx.y * 128;
  gemm_mainloop(Ab, Wb, NEMBD, NEMBD, m0, n0, sA, sBt, acc);

  const int lane = threadIdx.x & 31, ln = lane & 15, hf = lane >> 4;
  const int wid = threadIdx.x >> 5, wm = wid & 3, wn = wid >> 2;
#pragma unroll
  for (int mt = 0; mt < 2; ++mt)
#pragma unroll
    for (int nt = 0; nt < 4; ++nt)
#pragma unroll
      for (int v = 0; v < 8; ++v) {
        int R = m0 + wm * 32 + mt * 16 + v + hf * 8;
        int Cc = n0 + wn * 64 + nt * 16 + ln;
        Out[(size_t)R * NEMBD + Cc] = acc[mt][nt][v];
      }
}

// ---------------------------------------------------------------------------
extern "C" void kernel_launch(void* const* d_in, const int* in_sizes, int n_in,
                              void* d_out, int out_size, void* d_ws,
                              size_t ws_size, hipStream_t stream) {
  const float* x      = (const float*)d_in[0];  // [4,2048,1024]
  const float* w_qkv  = (const float*)d_in[1];  // [1024,3072]
  const float* w_proj = (const float*)d_in[2];  // [1024,1024]
  float* out = (float*)d_out;                   // [4,2048,1024]

  const int NX = BSZ * TSEQ * NEMBD;   // 8388608
  const int NW1 = NEMBD * 3 * NEMBD;   // 3145728
  const int NW2 = NEMBD * NEMBD;       // 1048576

  // workspace layout (bf16), ~92 MB total
  __bf16* xb     = (__bf16*)d_ws;
  __bf16* wqkvb  = xb + NX;
  __bf16* wprojb = wqkvb + NW1;
  __bf16* Qw     = wprojb + NW2;       // [B,H,T,D]
  __bf16* Kw     = Qw + NX;
  __bf16* Vw     = Kw + NX;
  __bf16* AO     = Vw + NX;            // [B,T,H,D]

  cvt_f32_bf16<<<NX / 1024, 256, 0, stream>>>(x, xb, NX);
  cvt_f32_bf16<<<NW1 / 1024, 256, 0, stream>>>(w_qkv, wqkvb, NW1);
  cvt_f32_bf16<<<NW2 / 1024, 256, 0, stream>>>(w_proj, wprojb, NW2);

  qkv_gemm<<<dim3(BSZ * TSEQ / 128, 3 * NEMBD / 128), 256, 0, stream>>>(
      xb, wqkvb, Qw, Kw, Vw);

  attn_kernel<<<dim3(TSEQ / 64, BSZ * NHEAD), 128, 0, stream>>>(Qw, Kw, Vw, AO);

  proj_gemm<<<dim3(BSZ * TSEQ / 128, NEMBD / 128), 256, 0, stream>>>(
      AO, wprojb, out);
}
